// EncoderLayer_59992103191047
// MI455X (gfx1250) — matrix operand order repaired and checked by reading, not compiled, not run on hardware
//
#include <hip/hip_runtime.h>
#include <math.h>

#define BDIM 4
#define LDIM 1024
#define KNB  30
#define DDIM 256
#define HDIM 8
#define DKD  32
#define BL   (BDIM*LDIM)        // 4096
#define NEDGE (BDIM*LDIM*KNB)   // 122880
#define D3   (3*DDIM)           // 768

typedef __attribute__((ext_vector_type(16))) _Float16 v16h;
typedef __attribute__((ext_vector_type(8)))  float    v8f;

// ---- WMMA fragment helpers (CDNA5 wave32 16x16x32 f16 layouts) ----
// A (16xK tile): lane&15 = row M; lane>>4 selects K-half; VGPR j holds K pair.
__device__ __forceinline__ v16h frag_a_lds_h(const _Float16* src, int stride, int m, int hi, int k0){
  v16h a;
#pragma unroll
  for(int j=0;j<8;++j){
    int kk = k0 + 2*j + 16*hi;
    a[2*j]   = src[m*stride + kk];
    a[2*j+1] = src[m*stride + kk + 1];
  }
  return a;
}
__device__ __forceinline__ v16h frag_a_lds_f(const float* src, int stride, int m, int hi, int k0){
  v16h a;
#pragma unroll
  for(int j=0;j<8;++j){
    int kk = k0 + 2*j + 16*hi;
    a[2*j]   = (_Float16)src[m*stride + kk];
    a[2*j+1] = (_Float16)src[m*stride + kk + 1];
  }
  return a;
}
// B (Kx16 tile) from an N-major (transposed) f16 weight: lane&15 = col N,
// lane>>4 selects K-half, slot e = consecutive K -> contiguous per-lane loads.
__device__ __forceinline__ v16h frag_b_nmajor(const _Float16* wt, int Kdim, int n, int hi, int k0){
  v16h b;
  const _Float16* p = wt + (size_t)n*Kdim + k0 + 16*hi;
#pragma unroll
  for(int e=0;e<16;++e) b[e] = p[e];
  return b;
}
__device__ __forceinline__ v8f wmma16(v16h a, v16h b, v8f c){
  return __builtin_amdgcn_wmma_f32_16x16x32_f16(false, a, false, b, (short)0, c, false, false);
}
__device__ __forceinline__ float gelu_exact(float x){
  return 0.5f * x * (1.0f + erff(x * 0.70710678118654752f));
}

// ---- small prep kernels ----
__global__ void wcvt_kernel(const float* __restrict__ src, _Float16* __restrict__ dst, int Kdim, int N){
  int total = Kdim*N;
  for(int i = blockIdx.x*blockDim.x + threadIdx.x; i < total; i += gridDim.x*blockDim.x){
    int k = i / N, n = i % N;
    dst[(size_t)n*Kdim + k] = (_Float16)src[i];   // store N-major (transposed)
  }
}

__global__ void bias_base_kernel(const float* __restrict__ nonE, const float* __restrict__ biasW,
                                 const float* __restrict__ biasB, float* __restrict__ base){
  __shared__ float red[256];
  float s = 0.f;
  for(int i = threadIdx.x; i < DDIM; i += 256) s += nonE[i]*biasW[i];
  red[threadIdx.x] = s;
  __syncthreads();
  for(int off=128; off>0; off>>=1){
    if(threadIdx.x < off) red[threadIdx.x] += red[threadIdx.x+off];
    __syncthreads();
  }
  if(threadIdx.x==0) base[0] = red[0] + biasB[0];
}

__global__ void bias_fill_kernel(float* __restrict__ bias, const float* __restrict__ base){
  float v = base[0];
  int total = BDIM*LDIM*LDIM;
  for(int i = blockIdx.x*blockDim.x + threadIdx.x; i < total; i += gridDim.x*blockDim.x) bias[i] = v;
}

// one thread per (b,l); k processed in order -> deterministic duplicate handling
__global__ void bias_scatter_kernel(const float* __restrict__ E, const int* __restrict__ Eidx,
                                    const float* __restrict__ biasW, const float* __restrict__ biasB,
                                    float* __restrict__ bias){
  int r = blockIdx.x*blockDim.x + threadIdx.x;     // (b*L + l)
  if(r >= BL) return;
  float bb = biasB[0];
  for(int k=0;k<KNB;++k){
    const float* er = E + ((size_t)r*KNB + k)*DDIM;
    float s = 0.f;
    for(int d=0; d<DDIM; ++d) s += er[d]*biasW[d];
    int j = Eidx[(size_t)r*KNB + k];
    bias[(size_t)r*LDIM + j] = s + bb;
  }
}

// ---- QKV projection: grid (BL/16, 3); out layout [b*H+h][l][dk] f16 ----
__global__ void qkv_kernel(const float* __restrict__ hV,
                           const _Float16* __restrict__ WqT, const _Float16* __restrict__ WkT,
                           const _Float16* __restrict__ WvT,
                           const float* __restrict__ bq, const float* __restrict__ bk,
                           const float* __restrict__ bv,
                           _Float16* __restrict__ qh, _Float16* __restrict__ kh, _Float16* __restrict__ vh){
  __shared__ _Float16 As[16*DDIM];
  const int tid = threadIdx.x, lane = tid & 31, wave = tid >> 5;
  const int m = lane & 15, hi = lane >> 4;
  const int row0 = blockIdx.x * 16;
  for(int i=tid; i<16*DDIM; i+=256)
    As[i] = (_Float16)hV[(size_t)(row0 + (i>>8))*DDIM + (i&255)];
  __syncthreads();
  const _Float16* WT = (blockIdx.y==0) ? WqT : ((blockIdx.y==1) ? WkT : WvT);
  const float*    bb = (blockIdx.y==0) ? bq  : ((blockIdx.y==1) ? bk  : bv);
  _Float16*      out = (blockIdx.y==0) ? qh  : ((blockIdx.y==1) ? kh  : vh);
#pragma unroll
  for(int t=0;t<2;++t){
    int n0 = (wave*2+t)*16;
    v8f c = {};
    for(int kc=0; kc<DDIM/32; ++kc){
      v16h a = frag_a_lds_h(As, DDIM, m, hi, kc*32);
      v16h b = frag_b_nmajor(WT, DDIM, n0 + m, hi, kc*32);
      c = wmma16(a,b,c);
    }
#pragma unroll
    for(int r=0;r<8;++r){
      int row = r + 8*hi, col = n0 + m;
      int gr = row0 + row;
      float val = c[r] + bb[col];
      int bidx = gr >> 10, l = gr & 1023;
      int h = col >> 5, dk = col & 31;
      out[(((size_t)(bidx*HDIM + h))*LDIM + l)*DKD + dk] = (_Float16)val;
    }
  }
}

// ---- Attention: grid = B*H*(L/16); 16 queries x 1024 keys per block ----
__global__ void attn_kernel(const _Float16* __restrict__ qh, const _Float16* __restrict__ kh,
                            const _Float16* __restrict__ vh, const float* __restrict__ bias,
                            const float* __restrict__ mask, float* __restrict__ ctx){
  __shared__ float    Slog[16*LDIM];   // 64 KB logits / probabilities
  __shared__ _Float16 Qt[16*DKD];
  __shared__ float    parts[8*16*16];
  __shared__ float    rmax[16], rsum[16];
  const int tid = threadIdx.x, lane = tid & 31, wave = tid >> 5;
  const int m = lane & 15, hi = lane >> 4;
  const int blk = blockIdx.x;
  const int qt = blk & 63;
  const int bh = blk >> 6;
  const int h = bh & 7, b = bh >> 3;
  const size_t hoff = (size_t)bh * LDIM * DKD;

  for(int i=tid; i<16*DKD; i+=256)
    Qt[i] = qh[hoff + (size_t)(qt*16 + (i>>5))*DKD + (i&31)];
  __syncthreads();

  {
    v16h aq = frag_a_lds_h(Qt, DKD, m, hi, 0);       // DK=32 -> single K-chunk
    const float scale = 0.17677669529663687f;        // 1/sqrt(32)
    for(int i=0;i<8;++i){
      int key0 = (wave*8 + i)*16;
      v16h bf;                                       // B = k^T: lane = key col
      const _Float16* kp = kh + hoff + (size_t)(key0 + m)*DKD + 16*hi;
#pragma unroll
      for(int e=0;e<16;++e) bf[e] = kp[e];
      v8f c = {};
      c = wmma16(aq, bf, c);
#pragma unroll
      for(int r=0;r<8;++r){
        int row = r + 8*hi, key = key0 + m;
        float val = c[r]*scale + bias[((size_t)b*LDIM + qt*16 + row)*LDIM + key];
        if(mask[b*LDIM + key] <= 0.f) val = -1e9f;
        Slog[row*LDIM + key] = val;
      }
    }
  }
  __syncthreads();
  if(tid < 16){
    float mx = -3.4e38f;
    for(int k=0;k<LDIM;++k) mx = fmaxf(mx, Slog[tid*LDIM + k]);
    rmax[tid] = mx;
  }
  __syncthreads();
  for(int i=tid; i<16*LDIM; i+=256){
    int row = i >> 10;
    Slog[i] = expf(Slog[i] - rmax[row]);
  }
  __syncthreads();
  if(tid < 16){
    float s = 0.f;
    for(int k=0;k<LDIM;++k) s += Slog[tid*LDIM + k];
    rsum[tid] = s;
  }
  __syncthreads();
  // attn @ v: N split over 2 tiles (DK=32), K=1024 split over 4 wave-groups
  {
    int nt = wave & 1, g = wave >> 1;
    v8f c = {};
    for(int i=0;i<8;++i){
      int k0 = (g*8 + i)*32;
      v16h a = frag_a_lds_f(Slog, LDIM, m, hi, k0);
      v16h bf;                                       // B = v: lane = dk col
      const _Float16* vp = vh + hoff + (size_t)(k0 + 16*hi)*DKD + nt*16 + m;
#pragma unroll
      for(int e=0;e<16;++e) bf[e] = vp[(size_t)e*DKD];
      c = wmma16(a, bf, c);
    }
#pragma unroll
    for(int r=0;r<8;++r)
      parts[wave*256 + (r + 8*hi)*16 + m] = c[r];
  }
  __syncthreads();
  for(int idx=tid; idx<16*DKD; idx+=256){
    int row = idx >> 5, dk = idx & 31;
    int ntt = dk >> 4, nn = dk & 15;
    float s = 0.f;
#pragma unroll
    for(int gg=0; gg<4; ++gg) s += parts[(gg*2 + ntt)*256 + row*16 + nn];
    s /= rsum[row];
    ctx[((size_t)b*LDIM + qt*16 + row)*DDIM + h*DKD + dk] = s;
  }
}

// ---- Output projection + residual + LN1 + mask -> h_V_new ----
__global__ void oproj_kernel(const float* __restrict__ ctx, const _Float16* __restrict__ WoT,
                             const float* __restrict__ bo, const float* __restrict__ hV,
                             const float* __restrict__ g1, const float* __restrict__ b1,
                             const float* __restrict__ mask, float* __restrict__ hVnew){
  __shared__ _Float16 As[16*DDIM];
  __shared__ float    Cs[16*DDIM];
  __shared__ float    mu[16], rstd[16];
  const int tid = threadIdx.x, lane = tid & 31, wave = tid >> 5;
  const int m = lane & 15, hi = lane >> 4;
  const int row0 = blockIdx.x * 16;
  for(int i=tid; i<16*DDIM; i+=256)
    As[i] = (_Float16)ctx[(size_t)(row0 + (i>>8))*DDIM + (i&255)];
  __syncthreads();
#pragma unroll
  for(int t=0;t<2;++t){
    int n0 = (wave*2+t)*16;
    v8f c = {};
    for(int kc=0; kc<DDIM/32; ++kc){
      v16h a = frag_a_lds_h(As, DDIM, m, hi, kc*32);
      v16h b = frag_b_nmajor(WoT, DDIM, n0 + m, hi, kc*32);
      c = wmma16(a,b,c);
    }
#pragma unroll
    for(int r=0;r<8;++r){
      int row = r + 8*hi, col = n0 + m;
      Cs[row*DDIM + col] = c[r] + bo[col] + hV[(size_t)(row0+row)*DDIM + col];
    }
  }
  __syncthreads();
  if(tid < 16){
    float s=0.f, s2=0.f;
    for(int d=0; d<DDIM; ++d){ float v = Cs[tid*DDIM+d]; s += v; s2 += v*v; }
    float mean = s/DDIM, var = s2/DDIM - mean*mean;
    mu[tid] = mean; rstd[tid] = rsqrtf(var + 1e-5f);
  }
  __syncthreads();
  for(int i=tid; i<16*DDIM; i+=256){
    int row = i >> 8, col = i & 255;
    int gr = row0 + row;
    float v = (Cs[i]-mu[row])*rstd[row]*g1[col] + b1[col];
    hVnew[(size_t)gr*DDIM + col] = v * mask[gr];
  }
}

// ---- Fused edge MLP (3 GEMMs + GELU + residual + LN2) -> h_E ----
__global__ void mlp_kernel(const float* __restrict__ hVnew, const float* __restrict__ E,
                           const int* __restrict__ Eidx,
                           const _Float16* __restrict__ W11T, const float* __restrict__ b11,
                           const _Float16* __restrict__ W12T, const float* __restrict__ b12,
                           const _Float16* __restrict__ W13T, const float* __restrict__ b13,
                           const float* __restrict__ g2, const float* __restrict__ bln2,
                           float* __restrict__ hE){
  __shared__ _Float16 A0[16*D3];      // h_EV tile (f16), reused as A2
  __shared__ float    Etile[16*DDIM]; // E residual (f32)
  __shared__ _Float16 A1[16*DDIM];
  __shared__ float    C3[16*DDIM];
  __shared__ int rb[16], rl[16], rk[16], rn[16];
  __shared__ float mu[16], rstd[16];
  const int tid = threadIdx.x, lane = tid & 31, wave = tid >> 5;
  const int m = lane & 15, hi = lane >> 4;
  const int e0 = blockIdx.x * 16;
  if(tid < 16){
    int e = e0 + tid;
    int b = e / (LDIM*KNB);
    int rem = e % (LDIM*KNB);
    rb[tid] = b; rl[tid] = rem / KNB; rk[tid] = rem % KNB;
    rn[tid] = Eidx[e];
  }
  __syncthreads();
  for(int i=tid; i<16*D3; i+=256){
    int r = i / D3, c = i % D3;
    float v;
    if(c < DDIM){
      v = hVnew[((size_t)rb[r]*LDIM + rl[r])*DDIM + c];
    } else if(c < 2*DDIM){
      v = E[(size_t)(e0 + r)*DDIM + (c - DDIM)];
      Etile[r*DDIM + (c - DDIM)] = v;
    } else {
      v = hVnew[((size_t)rb[r]*LDIM + rn[r])*DDIM + (c - 2*DDIM)];
    }
    A0[i] = (_Float16)v;
  }
  __syncthreads();
  // GEMM1 (K=768) + GELU -> A1
#pragma unroll
  for(int t=0;t<2;++t){
    int n0 = (wave*2+t)*16;
    v8f c = {};
    for(int kc=0; kc<D3/32; ++kc){
      __builtin_prefetch(W11T + (size_t)(n0+m)*D3 + kc*32 + 64, 0, 0);
      v16h a = frag_a_lds_h(A0, D3, m, hi, kc*32);
      v16h b = frag_b_nmajor(W11T, D3, n0 + m, hi, kc*32);
      c = wmma16(a,b,c);
    }
#pragma unroll
    for(int r=0;r<8;++r){
      int row = r + 8*hi, col = n0 + m;
      A1[row*DDIM + col] = (_Float16)gelu_exact(c[r] + b11[col]);
    }
  }
  __syncthreads();
  // GEMM2 (K=256) + GELU -> A2 (reuse A0 storage)
  _Float16* A2 = A0;
#pragma unroll
  for(int t=0;t<2;++t){
    int n0 = (wave*2+t)*16;
    v8f c = {};
    for(int kc=0; kc<DDIM/32; ++kc){
      v16h a = frag_a_lds_h(A1, DDIM, m, hi, kc*32);
      v16h b = frag_b_nmajor(W12T, DDIM, n0 + m, hi, kc*32);
      c = wmma16(a,b,c);
    }
#pragma unroll
    for(int r=0;r<8;++r){
      int row = r + 8*hi, col = n0 + m;
      A2[row*DDIM + col] = (_Float16)gelu_exact(c[r] + b12[col]);
    }
  }
  __syncthreads();
  // GEMM3 (K=256) + bias + residual -> C3
#pragma unroll
  for(int t=0;t<2;++t){
    int n0 = (wave*2+t)*16;
    v8f c = {};
    for(int kc=0; kc<DDIM/32; ++kc){
      v16h a = frag_a_lds_h(A2, DDIM, m, hi, kc*32);
      v16h b = frag_b_nmajor(W13T, DDIM, n0 + m, hi, kc*32);
      c = wmma16(a,b,c);
    }
#pragma unroll
    for(int r=0;r<8;++r){
      int row = r + 8*hi, col = n0 + m;
      C3[row*DDIM + col] = c[r] + b13[col] + Etile[row*DDIM + col];
    }
  }
  __syncthreads();
  if(tid < 16){
    float s=0.f, s2=0.f;
    for(int d=0; d<DDIM; ++d){ float v = C3[tid*DDIM+d]; s += v; s2 += v*v; }
    float mean = s/DDIM, var = s2/DDIM - mean*mean;
    mu[tid] = mean; rstd[tid] = rsqrtf(var + 1e-5f);
  }
  __syncthreads();
  for(int i=tid; i<16*DDIM; i+=256){
    int row = i >> 8, col = i & 255;
    hE[(size_t)(e0 + row)*DDIM + col] = (C3[i]-mu[row])*rstd[row]*g2[col] + bln2[col];
  }
}

extern "C" void kernel_launch(void* const* d_in, const int* in_sizes, int n_in,
                              void* d_out, int out_size, void* d_ws, size_t ws_size,
                              hipStream_t stream) {
  const float* hV   = (const float*)d_in[0];
  const float* E    = (const float*)d_in[1];
  const int*   Eidx = (const int*)  d_in[2];
  const float* mask = (const float*)d_in[3];
  const float* nonE = (const float*)d_in[4];
  const float* biasW= (const float*)d_in[5];
  const float* biasB= (const float*)d_in[6];
  const float* Wq = (const float*)d_in[7];  const float* bq = (const float*)d_in[8];
  const float* Wk = (const float*)d_in[9];  const float* bk = (const float*)d_in[10];
  const float* Wv = (const float*)d_in[11]; const float* bv = (const float*)d_in[12];
  const float* Wo = (const float*)d_in[13]; const float* bo = (const float*)d_in[14];
  const float* g1 = (const float*)d_in[15]; const float* b1 = (const float*)d_in[16];
  const float* W11= (const float*)d_in[17]; const float* b11= (const float*)d_in[18];
  const float* W12= (const float*)d_in[19]; const float* b12= (const float*)d_in[20];
  const float* W13= (const float*)d_in[21]; const float* b13= (const float*)d_in[22];
  const float* g2 = (const float*)d_in[23]; const float* b2 = (const float*)d_in[24];

  char* ws = (char*)d_ws;
  size_t off = 0;
  auto alloc = [&](size_t bytes)->void* {
    void* p = ws + off; off += (bytes + 255) & ~(size_t)255; return p;
  };
  float*    bias = (float*)   alloc((size_t)BDIM*LDIM*LDIM*sizeof(float)); // 16 MB
  float*    base = (float*)   alloc(256);
  _Float16* qh   = (_Float16*)alloc((size_t)BL*DDIM*sizeof(_Float16));
  _Float16* kh   = (_Float16*)alloc((size_t)BL*DDIM*sizeof(_Float16));
  _Float16* vh   = (_Float16*)alloc((size_t)BL*DDIM*sizeof(_Float16));
  float*    ctx  = (float*)   alloc((size_t)BL*DDIM*sizeof(float));
  _Float16* WqT  = (_Float16*)alloc((size_t)DDIM*DDIM*sizeof(_Float16));
  _Float16* WkT  = (_Float16*)alloc((size_t)DDIM*DDIM*sizeof(_Float16));
  _Float16* WvT  = (_Float16*)alloc((size_t)DDIM*DDIM*sizeof(_Float16));
  _Float16* WoT  = (_Float16*)alloc((size_t)DDIM*DDIM*sizeof(_Float16));
  _Float16* W11T = (_Float16*)alloc((size_t)D3*DDIM*sizeof(_Float16));
  _Float16* W12T = (_Float16*)alloc((size_t)DDIM*DDIM*sizeof(_Float16));
  _Float16* W13T = (_Float16*)alloc((size_t)DDIM*DDIM*sizeof(_Float16));

  float* hVnew = (float*)d_out;                       // (B,L,D)
  float* hE    = (float*)d_out + (size_t)BL*DDIM;     // (B,L,K,D)

  // weight precasts (f16, transposed to N-major)
  wcvt_kernel<<<64,256,0,stream>>>(Wq,  WqT,  DDIM, DDIM);
  wcvt_kernel<<<64,256,0,stream>>>(Wk,  WkT,  DDIM, DDIM);
  wcvt_kernel<<<64,256,0,stream>>>(Wv,  WvT,  DDIM, DDIM);
  wcvt_kernel<<<64,256,0,stream>>>(Wo,  WoT,  DDIM, DDIM);
  wcvt_kernel<<<64,256,0,stream>>>(W11, W11T, D3,   DDIM);
  wcvt_kernel<<<64,256,0,stream>>>(W12, W12T, DDIM, DDIM);
  wcvt_kernel<<<64,256,0,stream>>>(W13, W13T, DDIM, DDIM);

  // attention bias matrix
  bias_base_kernel<<<1,256,0,stream>>>(nonE, biasW, biasB, base);
  bias_fill_kernel<<<4096,256,0,stream>>>(bias, base);
  bias_scatter_kernel<<<(BL+255)/256,256,0,stream>>>(E, Eidx, biasW, biasB, bias);

  // MHA
  qkv_kernel<<<dim3(BL/16,3),256,0,stream>>>(hV, WqT,WkT,WvT, bq,bk,bv, qh,kh,vh);
  attn_kernel<<<BDIM*HDIM*(LDIM/16),256,0,stream>>>(qh,kh,vh, bias, mask, ctx);
  oproj_kernel<<<BL/16,256,0,stream>>>(ctx, WoT, bo, hV, g1, b1, mask, hVnew);

  // edge MLP (reads h_V_new from d_out, writes h_E)
  mlp_kernel<<<NEDGE/16,256,0,stream>>>(hVnew, E, Eidx, W11T,b11, W12T,b12, W13T,b13, g2,b2, hE);
}
